// TCABlock_46119358824533
// MI455X (gfx1250) — matrix operand order repaired and checked by reading, not compiled, not run on hardware
//
#include <hip/hip_runtime.h>
#include <hip/hip_bf16.h>
#include <stdint.h>
#include <stddef.h>

typedef __attribute__((ext_vector_type(16))) __bf16        v16bf;
typedef __attribute__((ext_vector_type(8)))  float         v8f;
typedef __attribute__((ext_vector_type(4)))  unsigned int  u32x4;
typedef __attribute__((ext_vector_type(8)))  int           i32x8;
typedef __attribute__((ext_vector_type(4)))  int           i32x4;

// generic-vector int4 matching the async builtin's parameter type exactly
typedef int v4i_g __attribute__((vector_size(16)));
typedef __attribute__((address_space(1))) v4i_g* as1_v4i;
typedef __attribute__((address_space(3))) v4i_g* as3_v4i;

#define K_DIM 1024
#define N_DIM 1024
#define M_DIM 32768
#define BM 128
#define BN 128
#define BK 32
#define LDT 40                     // halves per tile row in LDS (32 + 8 pad)
#define TILE_B (BM * LDT * 2)      // bytes per staged operand tile (10240)
#define NKSTEP (K_DIM / BK)

#if __has_builtin(__builtin_amdgcn_tensor_load_to_lds)
#define TDM_LDS 1
#elif __has_builtin(__builtin_amdgcn_global_load_async_to_lds_b128)
#define ASYNC_LDS 1
#endif

__device__ __forceinline__ unsigned short f2bf(float f) {
  unsigned int u = __float_as_uint(f);
  u += 0x7FFFu + ((u >> 16) & 1u);          // round-to-nearest-even
  return (unsigned short)(u >> 16);
}
__device__ __forceinline__ float bf2f(unsigned short h) {
  return __uint_as_float(((unsigned int)h) << 16);
}

__device__ __forceinline__ unsigned lds_off(const void* p) {
  return (unsigned)(uintptr_t)(__attribute__((address_space(3))) const void*)p;
}

#ifdef TDM_LDS
// One TDM descriptor: 2D tile [tile_dim1=128 rows x tile_dim0=32 elems] of a
// bf16 tensor with row stride 1024, padded in LDS 16B per 64B (LDT=40 halves).
// D# field packing per CDNA5 ISA.
__device__ __forceinline__ void tdm_load_tile(const unsigned short* gtile,
                                              unsigned ldsByte, unsigned rows) {
  unsigned long long ga = (unsigned long long)(uintptr_t)gtile;
  u32x4 g0;
  g0.x = 1u;                                  // count=1 (valid), user mode
  g0.y = ldsByte;                             // lds_addr
  g0.z = (unsigned)ga;                        // global_addr[31:0]
  g0.w = (unsigned)(ga >> 32) | (2u << 30);   // global_addr[56:32] | type=2
  i32x8 g1;
  g1[0] = (int)((1u << 16)                    // data_size = 2 bytes
              | (1u << 20)                    // pad_enable
              | (3u << 22)                    // pad_interval: 16 DWORDs (64B)
              | (3u << 25));                  // pad_amount: 4 DWORDs (16B)
  g1[1] = (int)(1024u << 16);                 // tensor_dim0 = 1024 (lo 16 @bit48)
  g1[2] = (int)((rows & 0xFFFFu) << 16);      // dim0 hi=0 | tensor_dim1 lo
  g1[3] = (int)((rows >> 16) | (32u << 16));  // tensor_dim1 hi | tile_dim0=32
  g1[4] = 128;                                // tile_dim1=128, tile_dim2=0
  g1[5] = 1024;                               // tensor_dim0_stride = 1024
  g1[6] = 0;                                  // stride hi | tensor_dim1_stride lo
  g1[7] = 0;
  i32x4 z4 = {0, 0, 0, 0};
  i32x8 z8 = {0, 0, 0, 0, 0, 0, 0, 0};
  __builtin_amdgcn_tensor_load_to_lds(g0, g1, z4, z4, z8, 0);
}
__device__ __forceinline__ void wait_tdm_all() {
#if __has_builtin(__builtin_amdgcn_s_wait_tensorcnt)
  __builtin_amdgcn_s_wait_tensorcnt(0);
#else
  asm volatile("s_wait_tensorcnt 0x0" ::: "memory");
#endif
}
#endif

__device__ __forceinline__ void cp16_to_lds(const unsigned short* g, unsigned short* l) {
#ifdef ASYNC_LDS
  __builtin_amdgcn_global_load_async_to_lds_b128(
      (as1_v4i)(uintptr_t)g, (as3_v4i)l, 0, 0);
#else
  *(u32x4*)l = *(const u32x4*)g;
#endif
}

__device__ __forceinline__ void wait_async_all() {
#ifdef ASYNC_LDS
#if __has_builtin(__builtin_amdgcn_s_wait_asynccnt)
  __builtin_amdgcn_s_wait_asynccnt(0);
#else
  asm volatile("s_wait_asynccnt 0x0" ::: "memory");
#endif
#endif
}

union FragU { u32x4 q[2]; v16bf v; };

// ---------------------------------------------------------------------------
// D[M,N] = (Ahi+Alo)[M,K] * (Bhi+Blo)^T[N,K] (+bias). All operands pre-split
// bf16; TDM double-buffered LDS staging + 24 WMMA / k-step.
// ---------------------------------------------------------------------------
__global__ __launch_bounds__(256) void gemm_split(
    const unsigned short* __restrict__ Ahi, const unsigned short* __restrict__ Alo,
    const unsigned short* __restrict__ Bhi, const unsigned short* __restrict__ Blo,
    const float* __restrict__ bias, float* __restrict__ D) {
  // [stage][array: Ahi,Alo,Bhi,Blo][128 rows * LDT halves] = 81920 B
  __shared__ __align__(16) unsigned short sbuf[2][4][BM * LDT];

  const int tid  = threadIdx.x;
  const int lane = tid & 31;
  const int wave = tid >> 5;
  const int wr = wave >> 2;            // 0..1 -> 64-row strip
  const int wc = wave & 3;             // 0..3 -> 32-col strip
  const int m0 = blockIdx.y * BM;
  const int n0 = blockIdx.x * BN;
  const int rowBase = wr * 64;
  const int colBase = wc * 32;
  const int l16 = lane & 15;
  const int akb = (lane < 16) ? 0 : 8;    // A frag K-base (ISA 16-bit A layout)
  const int bkb = (lane < 16) ? 0 : 8;    // B frag K-base (ISA 16-bit B layout)

#ifdef TDM_LDS
  // SGPR-uniform wave-0 test -> scalar branch, so waves 1..7 SKIP the tensor
  // ops entirely (TDM ignores EXEC; predication would issue 8x duplicates).
  const bool w0 = (__builtin_amdgcn_readfirstlane(wave) == 0);
  const unsigned ldsStage0 = lds_off(&sbuf[0][0][0]);
  const unsigned ldsStage1 = lds_off(&sbuf[1][0][0]);
  const unsigned short* gA_hi = Ahi + (size_t)m0 * K_DIM;
  const unsigned short* gA_lo = Alo + (size_t)m0 * K_DIM;
  const unsigned short* gB_hi = Bhi + (size_t)n0 * K_DIM;
  const unsigned short* gB_lo = Blo + (size_t)n0 * K_DIM;
  auto issue = [&](int i) {
    const int k0 = i * BK;
    const unsigned lb = (i & 1) ? ldsStage1 : ldsStage0;
    tdm_load_tile(gA_hi + k0, lb + 0 * TILE_B, M_DIM);
    tdm_load_tile(gA_lo + k0, lb + 1 * TILE_B, M_DIM);
    tdm_load_tile(gB_hi + k0, lb + 2 * TILE_B, N_DIM);
    tdm_load_tile(gB_lo + k0, lb + 3 * TILE_B, N_DIM);
  };
#else
  // per-lane staging: 64 threads per array, 8x16B chunks per thread
  const int arr = tid >> 6;               // 0..3
  const int u6  = tid & 63;
  const unsigned short* gsel =
      (arr == 0) ? Ahi : (arr == 1) ? Alo : (arr == 2) ? Bhi : Blo;
  const int rb = (arr < 2) ? m0 : n0;
  const unsigned short* gsrc = gsel + (size_t)rb * K_DIM;
  unsigned short* lbase0 = &sbuf[0][0][0];
  auto issue = [&](int i) {
    const int k0 = i * BK;
    unsigned short* lb = lbase0 + (size_t)(i & 1) * (4 * BM * LDT) + arr * (BM * LDT);
    #pragma unroll
    for (int j = 0; j < 8; ++j) {
      int c2  = u6 * 8 + j;               // 0..511
      int row = c2 >> 2;                  // 0..127
      int ch  = c2 & 3;                   // 16B chunk within row
      cp16_to_lds(gsrc + (size_t)row * K_DIM + k0 + ch * 8,
                  lb + row * LDT + ch * 8);
    }
  };
#endif

  const v8f vzero = {0.f,0.f,0.f,0.f,0.f,0.f,0.f,0.f};
  v8f acc[4][2];
  #pragma unroll
  for (int i = 0; i < 4; ++i)
    #pragma unroll
    for (int j = 0; j < 2; ++j) acc[i][j] = vzero;

#ifdef TDM_LDS
  if (w0) issue(0);
#else
  issue(0);
#endif

  for (int i = 0; i < NKSTEP; ++i) {
#ifdef TDM_LDS
    if (w0) wait_tdm_all();
#else
    wait_async_all();
#endif
    __syncthreads();                       // stage i resident; prior reads done
    const int cur = i & 1;
    const unsigned short* sAhi = sbuf[cur][0];
    const unsigned short* sAlo = sbuf[cur][1];
    const unsigned short* sBhi = sbuf[cur][2];
    const unsigned short* sBlo = sbuf[cur][3];

    // fragments (layouts per CDNA5 ISA §7.12.2)
    FragU ahi[4], alo[4], bhi[2], blo[2];
    #pragma unroll
    for (int rt = 0; rt < 4; ++rt) {
      int off = (rowBase + rt * 16 + l16) * LDT + akb;
      ahi[rt].q[0] = *(const u32x4*)(sAhi + off);
      ahi[rt].q[1] = *(const u32x4*)(sAhi + off + 16);
      alo[rt].q[0] = *(const u32x4*)(sAlo + off);
      alo[rt].q[1] = *(const u32x4*)(sAlo + off + 16);
    }
    #pragma unroll
    for (int ct = 0; ct < 2; ++ct) {
      int off = (colBase + ct * 16 + l16) * LDT + bkb;
      bhi[ct].q[0] = *(const u32x4*)(sBhi + off);
      bhi[ct].q[1] = *(const u32x4*)(sBhi + off + 16);
      blo[ct].q[0] = *(const u32x4*)(sBlo + off);
      blo[ct].q[1] = *(const u32x4*)(sBlo + off + 16);
    }

    if (i + 1 < NKSTEP) {                  // overlap next-stage copy with WMMA
#ifdef TDM_LDS
      if (w0) issue(i + 1);
#else
      issue(i + 1);
#endif
    }

    // split-bf16 fp32 emulation: hi*hi + hi*lo + lo*hi
    #pragma unroll
    for (int rt = 0; rt < 4; ++rt)
      #pragma unroll
      for (int ct = 0; ct < 2; ++ct) {
        acc[rt][ct] = __builtin_amdgcn_wmma_f32_16x16x32_bf16(
            false, ahi[rt].v, false, bhi[ct].v, (short)0, acc[rt][ct], false, false);
        acc[rt][ct] = __builtin_amdgcn_wmma_f32_16x16x32_bf16(
            false, ahi[rt].v, false, blo[ct].v, (short)0, acc[rt][ct], false, false);
        acc[rt][ct] = __builtin_amdgcn_wmma_f32_16x16x32_bf16(
            false, alo[rt].v, false, bhi[ct].v, (short)0, acc[rt][ct], false, false);
      }
  }

  // epilogue: C/D layout: VGPR r -> M = r + 8*(lane>=16), N = lane&15
  #pragma unroll
  for (int ct = 0; ct < 2; ++ct) {
    int col = n0 + colBase + ct * 16 + l16;
    float b = bias ? bias[col] : 0.f;
    #pragma unroll
    for (int rt = 0; rt < 4; ++rt) {
      int rbase = m0 + rowBase + rt * 16 + ((lane >> 4) << 3);
      #pragma unroll
      for (int r = 0; r < 8; ++r)
        D[(size_t)(rbase + r) * N_DIM + col] = acc[rt][ct][r] + b;
    }
  }
}

// ---------------------------------------------------------------------------
// Elementwise fp32 -> bf16 hi/lo split (for x and c). 8 elems/thread.
// ---------------------------------------------------------------------------
__global__ __launch_bounds__(256) void split_bf16(
    const float* __restrict__ src,
    unsigned short* __restrict__ hi, unsigned short* __restrict__ lo) {
  size_t base = ((size_t)blockIdx.x * 256 + threadIdx.x) * 8;
  float4 f0 = *(const float4*)(src + base);
  float4 f1 = *(const float4*)(src + base + 4);
  float ff[8] = {f0.x,f0.y,f0.z,f0.w,f1.x,f1.y,f1.z,f1.w};
  union { unsigned short s[8]; u32x4 v; } H, L;
  #pragma unroll
  for (int j = 0; j < 8; ++j) {
    unsigned short h = f2bf(ff[j]);
    H.s[j] = h;
    L.s[j] = f2bf(ff[j] - bf2f(h));
  }
  *(u32x4*)(hi + base) = H.v;
  *(u32x4*)(lo + base) = L.v;
}

// ---------------------------------------------------------------------------
// W[K,N] fp32 -> transposed bf16 hi/lo [N][K] (weight prep, done once).
// ---------------------------------------------------------------------------
__global__ __launch_bounds__(256) void splitT_bf16(
    const float* __restrict__ W,
    unsigned short* __restrict__ thi, unsigned short* __restrict__ tlo) {
  __shared__ float tile[32][33];
  const int tx = threadIdx.x & 31, ty = threadIdx.x >> 5;   // ty 0..7
  const int k0 = blockIdx.y * 32, n0 = blockIdx.x * 32;
  #pragma unroll
  for (int r = 0; r < 4; ++r)
    tile[ty * 4 + r][tx] = W[(size_t)(k0 + ty * 4 + r) * N_DIM + n0 + tx];
  __syncthreads();
  #pragma unroll
  for (int r = 0; r < 4; ++r) {
    int n = ty * 4 + r;
    float f = tile[tx][n];                 // = W[k0+tx][n0+n]
    unsigned short h = f2bf(f);
    size_t o = (size_t)(n0 + n) * K_DIM + k0 + tx;
    thi[o] = h;
    tlo[o] = f2bf(f - bf2f(h));
  }
}

// ---------------------------------------------------------------------------
// Per row i of [32768 x 1024] q/k/v: O(8x128) = softmax(Q K^T/sqrt(128)) V.
// (Raw .view reshape => each head unit is one contiguous 1024-float row.)
// Emits output pre-split as bf16 hi/lo for the final GEMM.
// ---------------------------------------------------------------------------
__global__ __launch_bounds__(256) void attn8(
    const float* __restrict__ q, const float* __restrict__ k,
    const float* __restrict__ v,
    unsigned short* __restrict__ ohi, unsigned short* __restrict__ olo) {
  __shared__ float qs[2][1024];
  __shared__ float ks[2][1024];
  __shared__ float vs[2][1024];
  __shared__ float S[2][64];

  const int t = threadIdx.x & 127;
  const int g = threadIdx.x >> 7;
  const size_t i = (size_t)blockIdx.x * 2 + g;
  const float* qp = q + i * 1024;
  const float* kp = k + i * 1024;
  const float* vp = v + i * 1024;

  #pragma unroll
  for (int j = 0; j < 8; ++j) {
    int idx = j * 128 + t;
    qs[g][idx] = qp[idx];
    ks[g][idx] = kp[idx];
    vs[g][idx] = vp[idx];
  }
  __syncthreads();

  if (t < 64) {
    int ii = t >> 3, jj = t & 7;
    const float* qr = &qs[g][ii * 128];
    const float* kr = &ks[g][jj * 128];
    float s = 0.f;
    #pragma unroll 8
    for (int c = 0; c < 128; ++c) s += qr[c] * kr[c];
    S[g][t] = s * 0.08838834764831845f;    // 1/sqrt(128) = scale^2
  }
  __syncthreads();

  if (t < 8) {
    float* row = &S[g][t * 8];
    float m = row[0];
    #pragma unroll
    for (int j = 1; j < 8; ++j) m = fmaxf(m, row[j]);
    float e[8], sum = 0.f;
    #pragma unroll
    for (int j = 0; j < 8; ++j) { e[j] = __expf(row[j] - m); sum += e[j]; }
    float inv = 1.f / sum;
    #pragma unroll
    for (int j = 0; j < 8; ++j) row[j] = e[j] * inv;
  }
  __syncthreads();

  #pragma unroll
  for (int r = 0; r < 8; ++r) {
    const float* P = &S[g][r * 8];
    float a = 0.f;
    #pragma unroll
    for (int j = 0; j < 8; ++j) a += P[j] * vs[g][j * 128 + t];
    unsigned short h = f2bf(a);
    size_t o = i * 1024 + r * 128 + t;
    ohi[o] = h;
    olo[o] = f2bf(a - bf2f(h));
  }
}

extern "C" void kernel_launch(void* const* d_in, const int* in_sizes, int n_in,
                              void* d_out, int out_size, void* d_ws, size_t ws_size,
                              hipStream_t stream) {
  (void)in_sizes; (void)n_in; (void)out_size; (void)ws_size;
  const float* x    = (const float*)d_in[0];
  const float* c    = (const float*)d_in[1];
  const float* Wq   = (const float*)d_in[2];
  const float* Wk   = (const float*)d_in[3];
  const float* Wv   = (const float*)d_in[4];
  const float* Wout = (const float*)d_in[5];
  const float* bout = (const float*)d_in[6];
  float* out = (float*)d_out;

  const size_t ME = (size_t)M_DIM * 1024;        // elements per [M,1024] buffer
  float* qb = (float*)d_ws;
  float* kb = qb + ME;
  float* vb = kb + ME;
  unsigned short* sh = (unsigned short*)(vb + ME);  // shared split-hi region
  unsigned short* sl = sh + ME;                     // shared split-lo region
  unsigned short* wq_h = sl + ME;                   // weight buffers: 8 x 1M halves
  unsigned short* wq_l = wq_h + (size_t)K_DIM * N_DIM;
  unsigned short* wk_h = wq_l + (size_t)K_DIM * N_DIM;
  unsigned short* wk_l = wk_h + (size_t)K_DIM * N_DIM;
  unsigned short* wv_h = wk_l + (size_t)K_DIM * N_DIM;
  unsigned short* wv_l = wv_h + (size_t)K_DIM * N_DIM;
  unsigned short* wo_h = wv_l + (size_t)K_DIM * N_DIM;
  unsigned short* wo_l = wo_h + (size_t)K_DIM * N_DIM;

  dim3 blk(256);
  dim3 gT(32, 32);                                  // weight transpose/split
  dim3 gG(N_DIM / BN, M_DIM / BM);                  // 8 x 256 GEMM grid
  unsigned gS = (unsigned)(ME / 2048);              // elementwise split
  unsigned gA = (unsigned)(M_DIM / 2);              // attention

  splitT_bf16<<<gT, blk, 0, stream>>>(Wq,   wq_h, wq_l);
  splitT_bf16<<<gT, blk, 0, stream>>>(Wk,   wk_h, wk_l);
  splitT_bf16<<<gT, blk, 0, stream>>>(Wv,   wv_h, wv_l);
  splitT_bf16<<<gT, blk, 0, stream>>>(Wout, wo_h, wo_l);

  split_bf16<<<gS, blk, 0, stream>>>(x, sh, sl);
  gemm_split<<<gG, blk, 0, stream>>>(sh, sl, wq_h, wq_l, (const float*)nullptr, qb);

  split_bf16<<<gS, blk, 0, stream>>>(c, sh, sl);
  gemm_split<<<gG, blk, 0, stream>>>(sh, sl, wk_h, wk_l, (const float*)nullptr, kb);
  gemm_split<<<gG, blk, 0, stream>>>(sh, sl, wv_h, wv_l, (const float*)nullptr, vb);

  attn8<<<gA, blk, 0, stream>>>(qb, kb, vb, sh, sl);   // region reuse
  gemm_split<<<gG, blk, 0, stream>>>(sh, sl, wo_h, wo_l, bout, out);
}
